// GCN_LSTM_27376121544892
// MI455X (gfx1250) — matrix operand order repaired and checked by reading, not compiled, not run on hardware
//
#include <hip/hip_runtime.h>

// ---------------------------------------------------------------------------
// GCN_LSTM for MI455X (gfx1250): bf16 WMMA GEMMs, LDS-resident LSTM state,
// register-resident LSTM weight fragments, async global->LDS tile staging,
// f32 atomics for graph aggregation.
// ---------------------------------------------------------------------------

#define N_NODES 1000
#define T_STEPS 128
#define F_IN    32
#define HID     64
#define G4      256           // 4*HID
#define NE      16000
#define M_TOT   (T_STEPS * N_NODES)   // 128000
#define D3      192

typedef __bf16 bf16_t;
typedef bf16_t bf16x16 __attribute__((ext_vector_type(16)));
typedef float  floatx8 __attribute__((ext_vector_type(8)));

__device__ __forceinline__ float sigmoidf_(float x) { return 1.0f / (1.0f + expf(-x)); }
__device__ __forceinline__ float leakyf_(float x)   { return x >= 0.0f ? x : 0.01f * x; }

// A-fragment K index map for 16x32 bf16 A tiles (contiguous: k = 16*half + 2*g).
__device__ __forceinline__ int a_kmap(int g, int half) {
    return 2 * g + 16 * half;
}

// Async copy of 16 bytes global -> LDS, tracked by ASYNCcnt.
__device__ __forceinline__ void async_copy_b128(void* lds_dst, const void* gsrc) {
    const unsigned lds_addr = (unsigned)(size_t)lds_dst;            // low 32 bits = LDS offset
    const unsigned long long ga = (unsigned long long)(size_t)gsrc;
    asm volatile("global_load_async_to_lds_b128 %0, %1, off"
                 :: "v"(lds_addr), "v"(ga) : "memory");
}
__device__ __forceinline__ void async_wait0() {
    asm volatile("s_wait_asynccnt 0x0" ::: "memory");
}

// ---------------------------------------------------------------------------
// Fused 2-layer LSTM over all T steps. One block owns 16 rows for the whole
// sequence; h/c live in LDS; weight fragments live in VGPRs.
// Layer1 K=96 -> 3 ksteps; Layer2 K=128 -> 4 ksteps.
// Writes node[t*N + n][64] = (bf16)leaky(h2).
// ---------------------------------------------------------------------------
__global__ __launch_bounds__(256) void lstm_full_kernel(
    const float* __restrict__ x,
    const float* __restrict__ Wih1, const float* __restrict__ Whh1,
    const float* __restrict__ bih1, const float* __restrict__ bhh1,
    const float* __restrict__ Wih2, const float* __restrict__ Whh2,
    const float* __restrict__ bih2, const float* __restrict__ bhh2,
    bf16_t* __restrict__ node)
{
    __shared__ __align__(16) bf16_t As[16 * 128];
    __shared__ float  Gs[16 * G4];
    __shared__ float  Hs1[16 * HID], Cs1[16 * HID], Hs2[16 * HID], Cs2[16 * HID];
    __shared__ float  Bias1[G4], Bias2[G4];

    const int tid  = threadIdx.x;
    const int wave = tid >> 5;
    const int lane = tid & 31;
    const int half = lane >> 4;
    const int ln16 = lane & 15;
    const int row0 = blockIdx.x * 16;

    for (int i = tid; i < 16 * HID; i += 256) { Hs1[i] = 0.f; Cs1[i] = 0.f; Hs2[i] = 0.f; Cs2[i] = 0.f; }
    for (int i = tid; i < G4; i += 256) { Bias1[i] = bih1[i] + bhh1[i]; Bias2[i] = bih2[i] + bhh2[i]; }

    // Preload B fragments (Wcat^T tiles) into registers; stay resident all T.
    bf16x16 bf1[2][3], bf2[2][4];
#pragma unroll
    for (int i = 0; i < 2; ++i) {
        const int j = (wave + 8 * i) * 16 + ln16;   // gate column
#pragma unroll
        for (int ks = 0; ks < 4; ++ks) {
#pragma unroll
            for (int g = 0; g < 8; ++g) {
                const int kb = ks * 32 + 16 * half + 2 * g;
                if (ks < 3) {
                    float v0 = (kb     < 32) ? Wih1[j * F_IN + kb]
                             : (kb     < 96) ? Whh1[j * HID + kb - 32] : 0.0f;
                    float v1 = (kb + 1 < 32) ? Wih1[j * F_IN + kb + 1]
                             : (kb + 1 < 96) ? Whh1[j * HID + kb + 1 - 32] : 0.0f;
                    bf1[i][ks][2 * g]     = (bf16_t)v0;
                    bf1[i][ks][2 * g + 1] = (bf16_t)v1;
                }
                float u0 = (kb     < 64) ? Wih2[j * HID + kb]     : Whh2[j * HID + kb - 64];
                float u1 = (kb + 1 < 64) ? Wih2[j * HID + kb + 1] : Whh2[j * HID + kb + 1 - 64];
                bf2[i][ks][2 * g]     = (bf16_t)u0;
                bf2[i][ks][2 * g + 1] = (bf16_t)u1;
            }
        }
    }
    __syncthreads();

    for (int t = 0; t < T_STEPS; ++t) {
        // ------- layer 1: A = [x_t | h1] (K=96; cols 96..127 unused) -------
        for (int idx = tid; idx < 16 * 128; idx += 256) {
            const int r = idx >> 7, k = idx & 127;
            const int n = row0 + r;
            float v = 0.0f;
            if (k < F_IN) {
                v = (n < N_NODES) ? x[((size_t)n * T_STEPS + t) * F_IN + k] : 0.0f;
                if (k == 0 && n < N_NODES && t + 1 < T_STEPS)
                    __builtin_prefetch(&x[((size_t)n * T_STEPS + (t + 1)) * F_IN], 0, 0);
            } else if (k < 96) {
                v = Hs1[r * HID + (k - F_IN)];
            }
            As[idx] = (bf16_t)v;
        }
        __syncthreads();
#pragma unroll
        for (int i = 0; i < 2; ++i) {
            floatx8 acc = {};
#pragma unroll
            for (int ks = 0; ks < 3; ++ks) {
                bf16x16 a;
#pragma unroll
                for (int g = 0; g < 8; ++g) {
                    const int k0 = a_kmap(g, half) + ks * 32;
                    a[2 * g]     = As[ln16 * 128 + k0];
                    a[2 * g + 1] = As[ln16 * 128 + k0 + 1];
                }
                acc = __builtin_amdgcn_wmma_f32_16x16x32_bf16(false, a, false, bf1[i][ks],
                                                              (short)0, acc, false, false);
            }
            const int col0 = (wave + 8 * i) * 16;
#pragma unroll
            for (int v = 0; v < 8; ++v)
                Gs[(v + 8 * half) * G4 + col0 + ln16] = acc[v];
        }
        __syncthreads();
        for (int idx = tid; idx < 16 * HID; idx += 256) {
            const int r = idx >> 6, hh = idx & 63;
            const float gi = sigmoidf_(Gs[r * G4 + hh]        + Bias1[hh]);
            const float gf = sigmoidf_(Gs[r * G4 + 64 + hh]   + Bias1[64 + hh]);
            const float gg = tanhf    (Gs[r * G4 + 128 + hh]  + Bias1[128 + hh]);
            const float go = sigmoidf_(Gs[r * G4 + 192 + hh]  + Bias1[192 + hh]);
            const float cn = gf * Cs1[idx] + gi * gg;
            Cs1[idx] = cn;
            Hs1[idx] = go * tanhf(cn);
        }
        __syncthreads();

        // ------- layer 2: A = [h1 | h2] (K=128) -------
        for (int idx = tid; idx < 16 * 128; idx += 256) {
            const int r = idx >> 7, k = idx & 127;
            const float v = (k < HID) ? Hs1[r * HID + k] : Hs2[r * HID + (k - HID)];
            As[idx] = (bf16_t)v;
        }
        __syncthreads();
#pragma unroll
        for (int i = 0; i < 2; ++i) {
            floatx8 acc = {};
#pragma unroll
            for (int ks = 0; ks < 4; ++ks) {
                bf16x16 a;
#pragma unroll
                for (int g = 0; g < 8; ++g) {
                    const int k0 = a_kmap(g, half) + ks * 32;
                    a[2 * g]     = As[ln16 * 128 + k0];
                    a[2 * g + 1] = As[ln16 * 128 + k0 + 1];
                }
                acc = __builtin_amdgcn_wmma_f32_16x16x32_bf16(false, a, false, bf2[i][ks],
                                                              (short)0, acc, false, false);
            }
            const int col0 = (wave + 8 * i) * 16;
#pragma unroll
            for (int v = 0; v < 8; ++v)
                Gs[(v + 8 * half) * G4 + col0 + ln16] = acc[v];
        }
        __syncthreads();
        for (int idx = tid; idx < 16 * HID; idx += 256) {
            const int r = idx >> 6, hh = idx & 63;
            const float gi = sigmoidf_(Gs[r * G4 + hh]        + Bias2[hh]);
            const float gf = sigmoidf_(Gs[r * G4 + 64 + hh]   + Bias2[64 + hh]);
            const float gg = tanhf    (Gs[r * G4 + 128 + hh]  + Bias2[128 + hh]);
            const float go = sigmoidf_(Gs[r * G4 + 192 + hh]  + Bias2[192 + hh]);
            const float cn = gf * Cs2[idx] + gi * gg;
            const float hn = go * tanhf(cn);
            Cs2[idx] = cn;
            Hs2[idx] = hn;
            const int n = row0 + r;
            if (n < N_NODES)
                node[((size_t)t * N_NODES + n) * HID + hh] = (bf16_t)leakyf_(hn);
        }
        __syncthreads();
    }
}

// ---------------------------------------------------------------------------
// GEMM: out[M x 64] = A[M x 64] @ W[64 x 64]; A, W in bf16, out f32.
// Tiles staged with GLOBAL_LOAD_ASYNC_TO_LDS_B128 (ASYNCcnt).
// ---------------------------------------------------------------------------
__global__ __launch_bounds__(256) void gemm64_kernel(
    const bf16_t* __restrict__ A, const bf16_t* __restrict__ W, float* __restrict__ out)
{
    __shared__ __align__(16) bf16_t As[64 * 64];
    __shared__ __align__(16) bf16_t Bs[64 * 64];
    const int tid  = threadIdx.x;
    const int wave = tid >> 5;
    const int lane = tid & 31;
    const int half = lane >> 4;
    const int ln16 = lane & 15;
    const size_t row0 = (size_t)blockIdx.x * 64;

    // 64x64 bf16 tile = 8192 B = 512 x b128; 256 threads x 2 per matrix.
#pragma unroll
    for (int i = 0; i < 2; ++i) {
        const int idx = tid + 256 * i;           // b128 slot
        async_copy_b128(As + idx * 8, A + row0 * 64 + (size_t)idx * 8);
        async_copy_b128(Bs + idx * 8, W + (size_t)idx * 8);
    }
    async_wait0();
    __syncthreads();

    const int mt = wave >> 1;
#pragma unroll
    for (int i = 0; i < 2; ++i) {
        const int ct = (wave & 1) * 2 + i;
        floatx8 acc = {};
#pragma unroll
        for (int ks = 0; ks < 2; ++ks) {
            bf16x16 a, b;
#pragma unroll
            for (int g = 0; g < 8; ++g) {
                const int k0 = a_kmap(g, half) + ks * 32;
                a[2 * g]     = As[(mt * 16 + ln16) * 64 + k0];
                a[2 * g + 1] = As[(mt * 16 + ln16) * 64 + k0 + 1];
                const int kb = ks * 32 + 16 * half + 2 * g;
                b[2 * g]     = Bs[kb * 64 + ct * 16 + ln16];
                b[2 * g + 1] = Bs[(kb + 1) * 64 + ct * 16 + ln16];
            }
            acc = __builtin_amdgcn_wmma_f32_16x16x32_bf16(false, a, false, b,
                                                          (short)0, acc, false, false);
        }
#pragma unroll
        for (int v = 0; v < 8; ++v)
            out[(row0 + mt * 16 + v + 8 * half) * 64 + ct * 16 + ln16] = acc[v];
    }
}

// Convert a 64x64 f32 weight block to bf16 (feeds async-staged GEMM).
__global__ void conv_w_kernel(const float* __restrict__ W, bf16_t* __restrict__ Wb) {
    const int idx = blockIdx.x * blockDim.x + threadIdx.x;
    if (idx < 64 * 64) Wb[idx] = (bf16_t)W[idx];
}

// ---------------------------------------------------------------------------
// Graph-norm helpers and aggregation
// ---------------------------------------------------------------------------
__global__ void zero_deg_kernel(float* __restrict__ deg) {
    const int n = blockIdx.x * blockDim.x + threadIdx.x;
    if (n < N_NODES) deg[n] = 0.0f;
}

__global__ void deg_count_kernel(const int* __restrict__ edges, float* __restrict__ deg) {
    const int e = blockIdx.x * blockDim.x + threadIdx.x;
    if (e < NE) atomicAdd(&deg[edges[NE + e]], 1.0f);  // dst row
}

__global__ void dis_kernel(const float* __restrict__ deg, float* __restrict__ dis) {
    const int n = blockIdx.x * blockDim.x + threadIdx.x;
    if (n < N_NODES) dis[n] = rsqrtf(deg[n] + 2.0f);
}

// agg = hW * self_norm + b
__global__ void agg_init_kernel(const float* __restrict__ hW, const float* __restrict__ dis,
                                const float* __restrict__ b, float* __restrict__ agg) {
    const int idx = blockIdx.x * blockDim.x + threadIdx.x;   // < M_TOT*64
    const int m = idx >> 6, d = idx & 63;
    const int n = m % N_NODES;
    const float s = 2.0f * dis[n] * dis[n];
    agg[idx] = hW[idx] * s + b[d];
}

// agg[t,dst,:] += hW[t,src,:] * dis[src]*dis[dst]  over all edges, timesteps
__global__ void agg_edges_kernel(const float* __restrict__ hW, const float* __restrict__ dis,
                                 const int* __restrict__ edges, float* __restrict__ agg) {
    const int idx = blockIdx.x * blockDim.x + threadIdx.x;   // < NE * T * 64
    const int e   = idx >> 13;                               // T*64 = 8192
    const int rem = idx & 8191;
    const int t   = rem >> 6;
    const int d   = rem & 63;
    const int src = edges[e];
    const int dst = edges[NE + e];
    const float w = dis[src] * dis[dst];
    atomicAdd(&agg[(t * N_NODES + dst) * 64 + d], hW[(t * N_NODES + src) * 64 + d] * w);
}

// hb = (bf16)leaky(agg)  -- inter-layer GCN activation feeding next bf16 GEMM
__global__ void leaky_bf_kernel(const float* __restrict__ a, bf16_t* __restrict__ hb) {
    const int idx = blockIdx.x * blockDim.x + threadIdx.x;   // < M_TOT*64
    hb[idx] = (bf16_t)leakyf_(a[idx]);
}

__global__ void leaky_concat_kernel(const float* __restrict__ a, float* __restrict__ cat, int coloff) {
    const int idx = blockIdx.x * blockDim.x + threadIdx.x;   // < M_TOT*64
    const int m = idx >> 6, d = idx & 63;
    cat[(size_t)m * D3 + coloff + d] = leakyf_(a[idx]);
}

// ---------------------------------------------------------------------------
// Compose three bias-linear layers (192->64->32->1) into weff[192], beff.
// ---------------------------------------------------------------------------
__global__ void compose_lin_kernel(const float* __restrict__ W1, const float* __restrict__ b1,
                                   const float* __restrict__ W2, const float* __restrict__ b2,
                                   const float* __restrict__ W3, const float* __restrict__ b3,
                                   float* __restrict__ weff, float* __restrict__ beff) {
    const int i = threadIdx.x;
    if (i < D3) {
        float acc = 0.0f;
        for (int j = 0; j < 32; ++j) {
            float s = 0.0f;
            for (int k = 0; k < 64; ++k) s += W1[i * 64 + k] * W2[k * 32 + j];
            acc += s * W3[j];
        }
        weff[i] = acc;
    }
    if (i == 0) {
        float be = b3[0];
        for (int j = 0; j < 32; ++j) {
            float s = b2[j];
            for (int k = 0; k < 64; ++k) s += b1[k] * W2[k * 32 + j];
            be += s * W3[j];
        }
        beff[0] = be;
    }
}

// out[n*T + t] = concat[t*N + n, :] . weff + beff    (one wave per row)
__global__ __launch_bounds__(256) void final_out_kernel(const float* __restrict__ cat,
                                                        const float* __restrict__ weff,
                                                        const float* __restrict__ beff,
                                                        float* __restrict__ out) {
    const int lane = threadIdx.x & 31;
    const int m = blockIdx.x * 8 + (threadIdx.x >> 5);
    if (m >= M_TOT) return;
    const int t = m / N_NODES;
    const int n = m % N_NODES;
    float s = 0.0f;
#pragma unroll
    for (int i = 0; i < D3 / 32; ++i) {
        const int d = lane + 32 * i;
        s += cat[(size_t)m * D3 + d] * weff[d];
    }
#pragma unroll
    for (int off = 16; off > 0; off >>= 1) s += __shfl_xor(s, off, 32);
    if (lane == 0) out[(size_t)n * T_STEPS + t] = s + beff[0];
}

// ---------------------------------------------------------------------------
// Host launcher
// ---------------------------------------------------------------------------
extern "C" void kernel_launch(void* const* d_in, const int* in_sizes, int n_in,
                              void* d_out, int out_size, void* d_ws, size_t ws_size,
                              hipStream_t stream) {
    (void)in_sizes; (void)n_in; (void)out_size; (void)ws_size;

    const float* x       = (const float*)d_in[0];
    const int*   e_stock = (const int*)d_in[1];
    const int*   e_to    = (const int*)d_in[2];
    const int*   e_from  = (const int*)d_in[3];
    // lstm params
    const float* Wih1 = (const float*)d_in[4];
    const float* Whh1 = (const float*)d_in[5];
    const float* bih1 = (const float*)d_in[6];
    const float* bhh1 = (const float*)d_in[7];
    const float* Wih2 = (const float*)d_in[8];
    const float* Whh2 = (const float*)d_in[9];
    const float* bih2 = (const float*)d_in[10];
    const float* bhh2 = (const float*)d_in[11];
    // gcn branches: (W1,b1,W2,b2) x3
    const float* gW[3][2] = {
        {(const float*)d_in[12], (const float*)d_in[14]},
        {(const float*)d_in[16], (const float*)d_in[18]},
        {(const float*)d_in[20], (const float*)d_in[22]}};
    const float* gB[3][2] = {
        {(const float*)d_in[13], (const float*)d_in[15]},
        {(const float*)d_in[17], (const float*)d_in[19]},
        {(const float*)d_in[21], (const float*)d_in[23]}};
    const int* edges[3] = {e_stock, e_to, e_from};
    // lin params
    const float* LW1 = (const float*)d_in[24];
    const float* Lb1 = (const float*)d_in[25];
    const float* LW2 = (const float*)d_in[26];
    const float* Lb2 = (const float*)d_in[27];
    const float* LW3 = (const float*)d_in[28];
    const float* Lb3 = (const float*)d_in[29];

    // workspace carve-up (byte offsets, 256B-aligned sections)
    char* wsb = (char*)d_ws;
    size_t off = 0;
    auto carve = [&](size_t bytes) { void* p = wsb + off; off += (bytes + 255) & ~(size_t)255; return p; };
    bf16_t* node_bf = (bf16_t*)carve((size_t)M_TOT * 64 * sizeof(bf16_t));
    bf16_t* hb      = (bf16_t*)carve((size_t)M_TOT * 64 * sizeof(bf16_t));
    bf16_t* wb      = (bf16_t*)carve((size_t)64 * 64 * sizeof(bf16_t));
    float*  hW      = (float*) carve((size_t)M_TOT * 64 * sizeof(float));
    float*  agg     = (float*) carve((size_t)M_TOT * 64 * sizeof(float));
    float*  cat     = (float*) carve((size_t)M_TOT * D3 * sizeof(float));
    float*  deg     = (float*) carve(1024 * sizeof(float));
    float*  dis     = (float*) carve(1024 * sizeof(float));
    float*  weff    = (float*) carve(256 * sizeof(float));
    float*  beff    = (float*) carve(16 * sizeof(float));

    // 1) LSTM (both layers, all timesteps) -> node = (bf16)leaky(h2)
    lstm_full_kernel<<<63, 256, 0, stream>>>(x, Wih1, Whh1, bih1, bhh1,
                                             Wih2, Whh2, bih2, bhh2, node_bf);

    const int ew_grid   = M_TOT * 64 / 256;        // 32000
    const int edge_grid = NE * T_STEPS * 64 / 256; // 512000
    const int gemm_grid = M_TOT / 64;              // 2000

    // 2) three GCN branches
    for (int b = 0; b < 3; ++b) {
        zero_deg_kernel<<<4, 256, 0, stream>>>(deg);
        deg_count_kernel<<<(NE + 255) / 256, 256, 0, stream>>>(edges[b], deg);
        dis_kernel<<<4, 256, 0, stream>>>(deg, dis);
        // layer 1
        conv_w_kernel<<<16, 256, 0, stream>>>(gW[b][0], wb);
        gemm64_kernel<<<gemm_grid, 256, 0, stream>>>(node_bf, wb, hW);
        agg_init_kernel<<<ew_grid, 256, 0, stream>>>(hW, dis, gB[b][0], agg);
        agg_edges_kernel<<<edge_grid, 256, 0, stream>>>(hW, dis, edges[b], agg);
        leaky_bf_kernel<<<ew_grid, 256, 0, stream>>>(agg, hb);
        // layer 2
        conv_w_kernel<<<16, 256, 0, stream>>>(gW[b][1], wb);
        gemm64_kernel<<<gemm_grid, 256, 0, stream>>>(hb, wb, hW);
        agg_init_kernel<<<ew_grid, 256, 0, stream>>>(hW, dis, gB[b][1], agg);
        agg_edges_kernel<<<edge_grid, 256, 0, stream>>>(hW, dis, edges[b], agg);
        leaky_concat_kernel<<<ew_grid, 256, 0, stream>>>(agg, cat, b * 64);
    }

    // 3) fold the 3 bias-linear layers into one vector, then dot per (t,n)
    compose_lin_kernel<<<1, 256, 0, stream>>>(LW1, Lb1, LW2, Lb2, LW3, Lb3, weff, beff);
    final_out_kernel<<<M_TOT / 8, 256, 0, stream>>>(cat, weff, beff, (float*)d_out);
}
